// SparseMatmul_4990751998153
// MI455X (gfx1250) — compile-verified, matrix operand order checked by reading, not run on hardware
//
#include <hip/hip_runtime.h>

#define M_DIM 8192
#define K_DIM 2048
#define N_DIM 2048
#define TILE 64
#define KTILES (K_DIM / TILE)   // 32
#define MTILES (M_DIM / TILE)   // 128
#define NTILES (N_DIM / TILE)   // 32
#define LDS_STRIDE 72           // 64 + pad: 144B row stride -> conflict-free-ish frag reads

typedef __attribute__((ext_vector_type(16))) __bf16 v16bf;
typedef __attribute__((ext_vector_type(8)))  float  v8f;

union FragAB { v16bf v; unsigned int u[8]; };
union FragC  { v8f   v; float        f[8]; };

// fp32 -> bf16 round-to-nearest-even (bit manipulation; no __bf16 arithmetic needed)
__device__ __forceinline__ unsigned short f2bf_rn(float f) {
  unsigned int u = __float_as_uint(f);
  unsigned int r = u + 0x7FFFu + ((u >> 16) & 1u);
  return (unsigned short)(r >> 16);
}
__device__ __forceinline__ float bf2f(unsigned short h) {
  return __uint_as_float(((unsigned int)h) << 16);
}

// ---------------------------------------------------------------------------
// Pass 1: per 64x64 tile nonzero flag for A.  grid = (KTILES, MTILES), 256 thr
// ---------------------------------------------------------------------------
__global__ void __launch_bounds__(256)
tile_flags_kernel(const float* __restrict__ a, int* __restrict__ flags) {
  __shared__ int nz;
  const int kt  = blockIdx.x;
  const int mt  = blockIdx.y;
  const int tid = threadIdx.x;
  if (tid == 0) nz = 0;
  __syncthreads();

  int local = 0;
  const float* tileBase = a + (size_t)(mt * TILE) * K_DIM + (size_t)kt * TILE;
#pragma unroll
  for (int it = 0; it < 4; ++it) {
    int idx = tid + it * 256;            // 0..1023 float4 slots (64 rows x 16)
    int row = idx >> 4;
    int col = (idx & 15) << 2;
    const float4 v = *(const float4*)(tileBase + (size_t)row * K_DIM + col);
    local |= (v.x != 0.f) | (v.y != 0.f) | (v.z != 0.f) | (v.w != 0.f);
  }
  if (local) atomicOr(&nz, 1);
  __syncthreads();
  if (tid == 0) flags[mt * KTILES + kt] = nz;
}

// ---------------------------------------------------------------------------
// Pass 2: tile-sparse GEMM. Block = 64x64 C tile, 8 wave32.
// Each wave: 16 rows x 32 cols of C = 2 f32 accumulators.
// Split precision: a = ah + al (bf16), b = bh + bl; acc += ah*bh + ah*bl + al*bh
// grid = (NTILES, MTILES), 256 threads
// ---------------------------------------------------------------------------
__global__ void __launch_bounds__(256)
sparse_wmma_gemm(const float* __restrict__ a, const float* __restrict__ b,
                 const int* __restrict__ flags, float* __restrict__ c) {
  __shared__ __align__(16) unsigned short sAhi[TILE * LDS_STRIDE];
  __shared__ __align__(16) unsigned short sAlo[TILE * LDS_STRIDE];
  __shared__ __align__(16) unsigned short sBhi[TILE * LDS_STRIDE];  // transposed [n][k]
  __shared__ __align__(16) unsigned short sBlo[TILE * LDS_STRIDE];

  const int nt   = blockIdx.x;
  const int mt   = blockIdx.y;
  const int tid  = threadIdx.x;
  const int lane = tid & 31;
  const int wave = tid >> 5;     // 0..7
  const int wm   = wave & 3;     // 16-row group within tile
  const int wn   = wave >> 2;    // 32-col group within tile
  const int half = lane >> 4;    // 0: lanes 0-15, 1: lanes 16-31
  const int lr   = lane & 15;

  FragC acc[2];
#pragma unroll
  for (int i = 0; i < 8; ++i) { acc[0].f[i] = 0.f; acc[1].f[i] = 0.f; }

  const int* frow = flags + mt * KTILES;

  for (int kt = 0; kt < KTILES; ++kt) {
    if (frow[kt] == 0) continue;         // block-uniform tile skip
    __syncthreads();                     // previous iter's frag reads complete

    // ---- A tile -> LDS (bf16 hi/lo), row-major [m][k]
    {
      const float* aBase = a + (size_t)(mt * TILE) * K_DIM + (size_t)kt * TILE;
#pragma unroll
      for (int it = 0; it < 4; ++it) {
        int idx = tid + it * 256;
        int row = idx >> 4;
        int col = (idx & 15) << 2;
        float4 v = *(const float4*)(aBase + (size_t)row * K_DIM + col);
        float vv[4] = {v.x, v.y, v.z, v.w};
#pragma unroll
        for (int e = 0; e < 4; ++e) {
          unsigned short hi = f2bf_rn(vv[e]);
          unsigned short lo = f2bf_rn(vv[e] - bf2f(hi));
          sAhi[row * LDS_STRIDE + col + e] = hi;
          sAlo[row * LDS_STRIDE + col + e] = lo;
        }
      }
      // ---- B tile -> LDS transposed [n][k] so K-pairs are contiguous
      const float* bBase = b + (size_t)(kt * TILE) * N_DIM + (size_t)nt * TILE;
#pragma unroll
      for (int it = 0; it < 4; ++it) {
        int idx  = tid + it * 256;
        int krow = idx >> 4;
        int ncol = (idx & 15) << 2;
        float4 v = *(const float4*)(bBase + (size_t)krow * N_DIM + ncol);
        float vv[4] = {v.x, v.y, v.z, v.w};
#pragma unroll
        for (int e = 0; e < 4; ++e) {
          unsigned short hi = f2bf_rn(vv[e]);
          unsigned short lo = f2bf_rn(vv[e] - bf2f(hi));
          sBhi[(ncol + e) * LDS_STRIDE + krow] = hi;
          sBlo[(ncol + e) * LDS_STRIDE + krow] = lo;
        }
      }
    }
    __syncthreads();

    // ---- two K=32 WMMA steps cover the 64-wide K tile
#pragma unroll
    for (int ks = 0; ks < 2; ++ks) {
      const int kb = ks * 32;

      // A fragment (16x32 bf16): lane lr -> row, K-pairs packed per dword,
      // lanes 16-31 hold K+8 / K+24 (ISA 16-bit A layout)
      FragAB ah, al;
      {
        const unsigned short* pHi = sAhi + (wm * 16 + lr) * LDS_STRIDE;
        const unsigned short* pLo = sAlo + (wm * 16 + lr) * LDS_STRIDE;
#pragma unroll
        for (int j = 0; j < 8; ++j) {
          int k = kb + (j >> 2) * 16 + half * 8 + (j & 3) * 2;
          ah.u[j] = *(const unsigned int*)(pHi + k);
          al.u[j] = *(const unsigned int*)(pLo + k);
        }
      }

#pragma unroll
      for (int nsub = 0; nsub < 2; ++nsub) {
        const int nb = wn * 32 + nsub * 16;
        FragAB bh, bl;
        {
          const unsigned short* pHi = sBhi + (nb + lr) * LDS_STRIDE;
          const unsigned short* pLo = sBlo + (nb + lr) * LDS_STRIDE;
#pragma unroll
          for (int j = 0; j < 8; ++j) {
            int k = kb + (j >> 2) * 16 + half * 8 + (j & 3) * 2;
            bh.u[j] = *(const unsigned int*)(pHi + k);
            bl.u[j] = *(const unsigned int*)(pLo + k);
          }
        }
        // error-compensated bf16 accumulation: ah*bh + ah*bl + al*bh
        acc[nsub].v = __builtin_amdgcn_wmma_f32_16x16x32_bf16(
            false, ah.v, false, bh.v, (short)0, acc[nsub].v, false, false);
        acc[nsub].v = __builtin_amdgcn_wmma_f32_16x16x32_bf16(
            false, ah.v, false, bl.v, (short)0, acc[nsub].v, false, false);
        acc[nsub].v = __builtin_amdgcn_wmma_f32_16x16x32_bf16(
            false, al.v, false, bh.v, (short)0, acc[nsub].v, false, false);
      }
    }
  }

  // ---- store C (16x16 f32 C/D layout: VGPR i -> row half*8+i, col = lane%16)
  const int rowBase = mt * TILE + wm * 16 + half * 8;
#pragma unroll
  for (int nsub = 0; nsub < 2; ++nsub) {
    const int colBase = nt * TILE + wn * 32 + nsub * 16;
    float* cp = c + (size_t)rowBase * N_DIM + colBase + lr;
#pragma unroll
    for (int i = 0; i < 8; ++i) {
      cp[(size_t)i * N_DIM] = acc[nsub].f[i];
    }
  }
}

extern "C" void kernel_launch(void* const* d_in, const int* in_sizes, int n_in,
                              void* d_out, int out_size, void* d_ws, size_t ws_size,
                              hipStream_t stream) {
  const float* a = (const float*)d_in[0];
  const float* b = (const float*)d_in[1];
  float* c       = (float*)d_out;
  int* flags     = (int*)d_ws;   // MTILES*KTILES ints = 16 KB

  dim3 fgrid(KTILES, MTILES);    // (32, 128)
  tile_flags_kernel<<<fgrid, 256, 0, stream>>>(a, flags);

  dim3 ggrid(NTILES, MTILES);    // (32, 128)
  sparse_wmma_gemm<<<ggrid, 256, 0, stream>>>(a, b, flags, c);
}
